// ResAttnBlock_14422500180420
// MI455X (gfx1250) — matrix operand order repaired and checked by reading, not compiled, not run on hardware
//
#include <hip/hip_runtime.h>
#include <hip/hip_bf16.h>

// ---------------------------------------------------------------------------
// ResAttnBlock for MI455X (gfx1250, wave32, WMMA bf16 16x16x32)
//   B=4, C=256, H=W=64 (N=4096), latent c=32
// Pipeline (all on `stream`, stream-ordered deps):
//   1. k_prep_xT      : x (B,C,N) f32 -> xT (B,N,C) bf16   (conv B-tiles)
//   2. k_qkv          : all six 1x1 convs (f32 math)
//   3. k_chan_energy  : energy_c (B,32,32) f32
//   4. k_chan_attn_M  : softmax + fold wo_c -> M (B,256,32) f32
//   5. k_sn           : one power iteration -> 1/sigma
//   6. k_wconv        : w_bar/sigma -> wh (9,256,256) bf16 (per-tap layout)
//   7. k_flash        : flash attention; DPP16 softmax reductions (no LDS
//                       bpermute stalls), V double-buffered, K prefetched,
//                       rescale skipped when running max unchanged
//   8. k_conv_final   : WMMA implicit-GEMM 3x3 conv, 2x2 subtiles + epilogue
// ---------------------------------------------------------------------------

typedef unsigned short u16;
typedef unsigned int   u32;
typedef __attribute__((ext_vector_type(16))) __bf16 v16bf;
typedef __attribute__((ext_vector_type(8)))  float  v8f;

#define NB   4
#define NC   256
#define NLAT 32
#define NPIX 4096   // 64*64
#define HW   64

union ABf { uint4 q[2]; v16bf v; };

__device__ __forceinline__ v8f zero8() {
  v8f z;
#pragma unroll
  for (int i = 0; i < 8; ++i) z[i] = 0.f;
  return z;
}

__device__ __forceinline__ v16bf zero16bf() {
  ABf u;
  u.q[0] = make_uint4(0u, 0u, 0u, 0u);
  u.q[1] = make_uint4(0u, 0u, 0u, 0u);
  return u.v;
}

// A-matrix (16x32 bf16) fragment load. rowptr = &row[kb], kb = (lane>=16)?8:0.
// VGPR0-3 <- k = kb..kb+7 ; VGPR4-7 <- k = kb+16..kb+23  (matches ISA layout)
__device__ __forceinline__ v16bf loadA(const u16* rowptr) {
  ABf u;
  u.q[0] = *(const uint4*)(rowptr);
  u.q[1] = *(const uint4*)(rowptr + 16);
  return u.v;
}

// B-matrix (32x16 bf16) fragment load from N-major storage.
// colptr = &col[kb2], kb2 = (lane>=16)?8:0 ; k = kb2..kb2+7, kb2+16..kb2+23.
__device__ __forceinline__ v16bf loadB32(const u16* colptr) {
  ABf u;
  u.q[0] = *(const uint4*)(colptr);
  u.q[1] = *(const uint4*)(colptr + 16);
  return u.v;
}

__device__ __forceinline__ v8f wmma_bf16(v16bf a, v16bf b, v8f c) {
  return __builtin_amdgcn_wmma_f32_16x16x32_bf16(false, a, false, b,
                                                 (short)0, c, false, false);
}

__device__ __forceinline__ u16 f2bf(float f) {  // round-to-nearest-even
  u32 x = __float_as_uint(f);
  u32 r = x + 0x7fffu + ((x >> 16) & 1u);
  return (u16)(r >> 16);
}

// ---- DPP16 16-lane butterfly reductions (pure VALU, no LDS / dscnt stalls).
// Sequence quad_perm[1,0,3,2] -> quad_perm[2,3,0,1] -> row_half_mirror ->
// row_mirror fully reduces each 16-lane half and broadcasts to all its lanes.
__device__ __forceinline__ float dpp_bfly(float v, int ctrl_is /*imm*/);

#define DPP_STEP(v, CTRL, OP)                                              \
  {                                                                        \
    int _y = __builtin_amdgcn_update_dpp(0, __float_as_int(v), (CTRL),     \
                                         0xF, 0xF, true);                  \
    (v) = OP((v), __int_as_float(_y));                                     \
  }

__device__ __forceinline__ float dpp_max16(float v) {
  DPP_STEP(v, 0xB1, fmaxf);   // quad_perm [1,0,3,2]
  DPP_STEP(v, 0x4E, fmaxf);   // quad_perm [2,3,0,1]
  DPP_STEP(v, 0x141, fmaxf);  // row_half_mirror
  DPP_STEP(v, 0x140, fmaxf);  // row_mirror
  return v;
}
__device__ __forceinline__ float addf(float a, float b) { return a + b; }
__device__ __forceinline__ float dpp_sum16(float v) {
  DPP_STEP(v, 0xB1, addf);
  DPP_STEP(v, 0x4E, addf);
  DPP_STEP(v, 0x141, addf);
  DPP_STEP(v, 0x140, addf);
  return v;
}

// ---------------------------------------------------------------- 1: x -> xT
__global__ __launch_bounds__(256) void k_prep_xT(const float* __restrict__ x,
                                                 u16* __restrict__ xT) {
  size_t i = (size_t)blockIdx.x * 256 + threadIdx.x;  // B*C*N elements
  int n = (int)(i & (NPIX - 1));
  int c = (int)((i >> 12) & (NC - 1));
  int b = (int)(i >> 20);
  xT[((size_t)b * NPIX + n) * NC + c] = f2bf(x[i]);
}

// ------------------------------------------------------- 2: all 1x1 convs
__global__ __launch_bounds__(256) void k_qkv(
    const float* __restrict__ x,
    const float* __restrict__ wq_fm, const float* __restrict__ bq_fm,
    const float* __restrict__ wk_fm, const float* __restrict__ bk_fm,
    const float* __restrict__ wv_fm, const float* __restrict__ bv_fm,
    const float* __restrict__ wq_c,  const float* __restrict__ bq_c,
    const float* __restrict__ wk_c,  const float* __restrict__ bk_c,
    const float* __restrict__ wv_c,  const float* __restrict__ bv_c,
    u16* __restrict__ qT, u16* __restrict__ kT, u16* __restrict__ vh,
    float* __restrict__ qc, float* __restrict__ kc, float* __restrict__ vc) {
  __shared__ float xs[NC][32];  // 32 pixels worth of x columns (32 KB)
  const int t  = threadIdx.x;
  const int p  = t & 31;        // pixel within tile == lane
  const int og = t >> 5;        // wave id == output group
  const int b  = blockIdx.x >> 7;            // NPIX/32 = 128 tiles per batch
  const int nb = (blockIdx.x & 127) * 32;
  const int n  = nb + p;

  for (int c = og; c < NC; c += 8)
    xs[c][p] = x[((size_t)b * NC + c) * NPIX + nb + p];
  __syncthreads();

  auto dotf = [&](const float* __restrict__ W, int o) -> float {
    float a = 0.f;
#pragma unroll 8
    for (int c = 0; c < NC; ++c) a = fmaf(W[o * NC + c], xs[c][p], a);
    return a;
  };

  for (int o = og; o < NLAT; o += 8) {
    float q = dotf(wq_fm, o) + bq_fm[o];
    float k = dotf(wk_fm, o) + bk_fm[o];
    qT[((size_t)b * NPIX + n) * NLAT + o] = f2bf(q);
    kT[((size_t)b * NPIX + n) * NLAT + o] = f2bf(k);
    qc[((size_t)b * NLAT + o) * NPIX + n] = dotf(wq_c, o) + bq_c[o];
    kc[((size_t)b * NLAT + o) * NPIX + n] = dotf(wk_c, o) + bk_c[o];
    vc[((size_t)b * NLAT + o) * NPIX + n] = dotf(wv_c, o) + bv_c[o];
  }
  for (int o = og; o < NC; o += 8) {
    float v = dotf(wv_fm, o) + bv_fm[o];
    vh[((size_t)b * NC + o) * NPIX + n] = f2bf(v);
  }
}

// ------------------------------------------------- 3: channel-attn energy
// block: 256 thr = 8 n-chunks x 32 j ; grid (B, 32 i)
__global__ __launch_bounds__(256) void k_chan_energy(
    const float* __restrict__ kc, const float* __restrict__ qc,
    float* __restrict__ energy) {
  __shared__ float red[8][32];
  const int b = blockIdx.x, i = blockIdx.y;
  const int j = threadIdx.x & 31;
  const int ch = threadIdx.x >> 5;
  const float4* ki = (const float4*)(kc + ((size_t)b * NLAT + i) * NPIX) + ch * 128;
  const float4* qj = (const float4*)(qc + ((size_t)b * NLAT + j) * NPIX) + ch * 128;
  float e = 0.f;
#pragma unroll 4
  for (int n = 0; n < 128; ++n) {
    float4 a = ki[n], c = qj[n];
    e += a.x * c.x + a.y * c.y + a.z * c.z + a.w * c.w;
  }
  red[ch][j] = e;
  __syncthreads();
  if (ch == 0) {
    float s = e;
#pragma unroll
    for (int k = 1; k < 8; ++k) s += red[k][j];
    energy[(b * NLAT + i) * NLAT + j] = s;
  }
}

// ------------------------------------- 4: softmax + fold wo_c into M(b,o,j)
__global__ __launch_bounds__(256) void k_chan_attn_M(
    const float* __restrict__ energy, const float* __restrict__ wo_c,
    float* __restrict__ Mmat) {
  __shared__ float attn[NLAT][NLAT];
  const int b = blockIdx.x, t = threadIdx.x;
  if (t < NLAT) {
    const float* er = energy + (b * NLAT + t) * NLAT;
    float mx = -1e30f;
    for (int j = 0; j < NLAT; ++j) mx = fmaxf(mx, er[j]);
    float s = 0.f, e[NLAT];
    for (int j = 0; j < NLAT; ++j) { e[j] = __expf(er[j] - mx); s += e[j]; }
    float inv = 1.f / s;
    for (int j = 0; j < NLAT; ++j) attn[t][j] = e[j] * inv;
  }
  __syncthreads();
  // M[b,o,j] = sum_i wo[o,i] * attn[j,i]
  float* Mo = Mmat + ((size_t)b * NC + t) * NLAT;
  for (int j = 0; j < NLAT; ++j) {
    float a = 0.f;
#pragma unroll
    for (int i = 0; i < NLAT; ++i) a = fmaf(wo_c[t * NLAT + i], attn[j][i], a);
    Mo[j] = a;
  }
}

// --------------------------------------------- 5: spectral norm (1/sigma)
__global__ __launch_bounds__(256) void k_sn(const float* __restrict__ w_bar,
                                            const float* __restrict__ u_sn,
                                            float* __restrict__ invsig) {
  __shared__ float t1s[2304];
  __shared__ float red[256];
  const int t = threadIdx.x;
  for (int k = t; k < 2304; k += 256) {
    float a = 0.f;
    for (int o = 0; o < NC; ++o) a = fmaf(w_bar[(size_t)o * 2304 + k], u_sn[o], a);
    t1s[k] = a;
  }
  __syncthreads();
  float ss = 0.f;
  for (int k = t; k < 2304; k += 256) ss += t1s[k] * t1s[k];
  red[t] = ss;
  __syncthreads();
  for (int s = 128; s > 0; s >>= 1) {
    if (t < s) red[t] += red[t + s];
    __syncthreads();
  }
  const float vn = sqrtf(red[0]) + 1e-12f;  // ||W^T u|| + eps
  __syncthreads();
  float so = 0.f;  // s[o] = (W t1)[o] / vn
  for (int k = 0; k < 2304; ++k) so = fmaf(w_bar[(size_t)t * 2304 + k], t1s[k], so);
  so /= vn;
  red[t] = so * so;
  __syncthreads();
  for (int s = 128; s > 0; s >>= 1) {
    if (t < s) red[t] += red[t + s];
    __syncthreads();
  }
  if (t == 0) {
    float ssum = red[0];
    float sigma = ssum / (sqrtf(ssum) + 1e-12f);  // u_new . s
    invsig[0] = 1.f / sigma;
  }
}

// --------------------------------- 6: scaled conv weights, per-tap (s,o,ci)
__global__ __launch_bounds__(256) void k_wconv(const float* __restrict__ w_bar,
                                               const float* __restrict__ invsig,
                                               u16* __restrict__ wh) {
  size_t i = (size_t)blockIdx.x * 256 + threadIdx.x;  // 256*256*9
  int o = (int)(i / 2304);
  int r = (int)(i % 2304);
  int ci = r / 9, s = r % 9;
  wh[((size_t)s * NC + o) * NC + ci] = f2bf(w_bar[i] * invsig[0]);
}

// ----------------------------------------------- 7: flash attention (WMMA)
// Block = 4 waves, one 32-row query tile per block. Every wave redundantly
// computes S + softmax (no barriers); each wave owns 64 of the 256 channels
// (4 acc tiles x 2 row-subtiles = 64 VGPRs) so V loads can be pipelined.
#define FL_WAVES 4
__global__ __launch_bounds__(FL_WAVES * 32) void k_flash(
    const u16* __restrict__ qT, const u16* __restrict__ kT,
    const u16* __restrict__ vh, const float* __restrict__ gamma_fm,
    float* __restrict__ fmbuf) {
  __shared__ u16  Pbuf[FL_WAVES][2][16][32];  // P tiles per wave/subtile
  __shared__ float stats[FL_WAVES][2][16];
  const int lane = threadIdx.x & 31;
  const int wv   = threadIdx.x >> 5;
  const int half = lane >> 4;
  const int l15  = lane & 15;
  const int b    = blockIdx.y;
  const int ib   = blockIdx.x * 32;   // 32 query rows per block
  const int ccg0 = wv * 4;            // this wave's first channel tile

  const u16* qTb = qT + ((size_t)b * NPIX + ib) * NLAT;
  const u16* kTb = kT + (size_t)b * NPIX * NLAT;
  const u16* vhb = vh + (size_t)b * NC * NPIX;

  const int kbA = half ? 8 : 0;
  const int kbB = half ? 8 : 0;
  v16bf Aq[2];
  Aq[0] = loadA(qTb + (size_t)l15 * NLAT + kbA);
  Aq[1] = loadA(qTb + (size_t)(16 + l15) * NLAT + kbA);

  float m[2][8], l[2][8];
  v8f acc[4][2];  // [channel tile][row subtile]
#pragma unroll
  for (int st = 0; st < 2; ++st)
#pragma unroll
    for (int r = 0; r < 8; ++r) { m[st][r] = -1e30f; l[st][r] = 0.f; }
#pragma unroll
  for (int cc = 0; cc < 4; ++cc) { acc[cc][0] = zero8(); acc[cc][1] = zero8(); }

  v16bf Bk0 = loadB32(kTb + (size_t)l15 * NLAT + kbB);
  v16bf Bk1 = loadB32(kTb + (size_t)(16 + l15) * NLAT + kbB);

  for (int j = 0; j < NPIX; j += 32) {
    v8f S[2][2];
    S[0][0] = wmma_bf16(Aq[0], Bk0, zero8());
    S[0][1] = wmma_bf16(Aq[0], Bk1, zero8());
    S[1][0] = wmma_bf16(Aq[1], Bk0, zero8());
    S[1][1] = wmma_bf16(Aq[1], Bk1, zero8());

    // streaming softmax per row (row = r + 8*half, cols across 16 lanes),
    // DPP16 butterfly reductions -- no LDS, no dscnt stalls.
    bool upd = false;
#pragma unroll
    for (int st = 0; st < 2; ++st)
#pragma unroll
      for (int r = 0; r < 8; ++r) {
        float mx = dpp_max16(fmaxf(S[st][0][r], S[st][1][r]));
        float mn = fmaxf(m[st][r], mx);
        float sc = __expf(m[st][r] - mn);
        float p0 = __expf(S[st][0][r] - mn);
        float p1 = __expf(S[st][1][r] - mn);
        float rs = dpp_sum16(p0 + p1);
        l[st][r] = l[st][r] * sc + rs;
        upd = upd || (mn > m[st][r]);
        m[st][r] = mn;
        const int il = r + 8 * half;
        Pbuf[wv][st][il][l15]      = f2bf(p0);
        Pbuf[wv][st][il][l15 + 16] = f2bf(p1);
        if (l15 == 0) stats[wv][st][il] = sc;  // lanes 0/16: rows 0-7 / 8-15
      }

    // prefetch next K fragments into registers (overlaps with PV section)
    if (j + 32 < NPIX) {
      Bk0 = loadB32(kTb + (size_t)(j + 32 + l15) * NLAT + kbB);
      Bk1 = loadB32(kTb + (size_t)(j + 48 + l15) * NLAT + kbB);
    }

    // rescale accumulators only when some row's max moved (scale != 1.0)
    if (__ballot(upd)) {
      const float scl0 = stats[wv][0][l15];
      const float scl1 = stats[wv][1][l15];
#pragma unroll
      for (int cc = 0; cc < 4; ++cc)
#pragma unroll
        for (int e = 0; e < 8; ++e) {
          acc[cc][0][e] *= scl0;
          acc[cc][1][e] *= scl1;
        }
    }

    // PV: acc[o,i] += V(o x 32j) * P^T(32j x i), V double-buffered
    const v16bf Bp0 = loadB32(&Pbuf[wv][0][l15][half * 8]);
    const v16bf Bp1 = loadB32(&Pbuf[wv][1][l15][half * 8]);
    v16bf Av = loadA(vhb + (size_t)(ccg0 * 16 + l15) * NPIX + j + kbA);
#pragma unroll
    for (int cc = 0; cc < 4; ++cc) {
      v16bf Avn = (cc < 3)
          ? loadA(vhb + (size_t)((ccg0 + cc + 1) * 16 + l15) * NPIX + j + kbA)
          : Av;
      acc[cc][0] = wmma_bf16(Av, Bp0, acc[cc][0]);
      acc[cc][1] = wmma_bf16(Av, Bp1, acc[cc][1]);
      Av = Avn;
    }
  }

#pragma unroll
  for (int st = 0; st < 2; ++st)
    if (l15 == 0)
#pragma unroll
      for (int r = 0; r < 8; ++r) stats[wv][st][r + 8 * half] = 1.f / l[st][r];
  const float linv0 = stats[wv][0][l15];
  const float linv1 = stats[wv][1][l15];
  const float g = *gamma_fm;
#pragma unroll
  for (int cc = 0; cc < 4; ++cc)
#pragma unroll
    for (int r = 0; r < 8; ++r) {
      int o = (ccg0 + cc) * 16 + r + 8 * half;
      size_t rowb = ((size_t)b * NC + o) * NPIX + ib;
      fmbuf[rowb + l15]      = g * acc[cc][0][r] * linv0;
      fmbuf[rowb + 16 + l15] = g * acc[cc][1][r] * linv1;
    }
}

// ------------------- 8: 3x3 conv (implicit GEMM, 2x2 subtiles) + epilogue
__global__ __launch_bounds__(128) void k_conv_final(
    const u16* __restrict__ xT, const u16* __restrict__ wh,
    const float* __restrict__ x, const float* __restrict__ fmbuf,
    const float* __restrict__ Mmat, const float* __restrict__ vc,
    const float* __restrict__ bo_c, const float* __restrict__ gamma_c_p,
    const float* __restrict__ res_gamma_p, float* __restrict__ out) {
  const int lane = threadIdx.x & 31;
  const int wv   = threadIdx.x >> 5;
  const int half = lane >> 4;
  const int l15  = lane & 15;
  const int b    = blockIdx.z;
  const int ob   = blockIdx.y * 32;                // 32 output channels
  const int pb   = (blockIdx.x * 4 + wv) * 32;     // 32 pixels
  const int kbA  = half ? 8 : 0;
  const int kbB  = half ? 8 : 0;

  const int p0 = pb + l15,      h0 = p0 >> 6, w0 = p0 & 63;
  const int p1 = pb + 16 + l15, h1 = p1 >> 6, w1 = p1 & 63;

  const u16* xb = xT + (size_t)b * NPIX * NC;
  v8f acc[2][2];  // [o subtile][p subtile]
  acc[0][0] = zero8(); acc[0][1] = zero8();
  acc[1][0] = zero8(); acc[1][1] = zero8();

#pragma unroll
  for (int kh = 0; kh < 3; ++kh)
#pragma unroll
    for (int kw = 0; kw < 3; ++kw) {
      const int dh = kh - 1, dw = kw - 1;
      const bool v0 = (h0 + dh >= 0) && (h0 + dh < HW) && (w0 + dw >= 0) && (w0 + dw < HW);
      const bool v1 = (h1 + dh >= 0) && (h1 + dh < HW) && (w1 + dw >= 0) && (w1 + dw < HW);
      const u16* colp0 = xb + (size_t)((h0 + dh) * HW + w0 + dw) * NC + kbB;
      const u16* colp1 = xb + (size_t)((h1 + dh) * HW + w1 + dw) * NC + kbB;
      const u16* wb0 = wh + ((size_t)(kh * 3 + kw) * NC + (ob + l15)) * NC + kbA;
      const u16* wb1 = wb0 + (size_t)16 * NC;
#pragma unroll
      for (int ci = 0; ci < NC; ci += 32) {
        v16bf Aw0 = loadA(wb0 + ci);
        v16bf Aw1 = loadA(wb1 + ci);
        v16bf Bx0 = v0 ? loadB32(colp0 + ci) : zero16bf();
        v16bf Bx1 = v1 ? loadB32(colp1 + ci) : zero16bf();
        acc[0][0] = wmma_bf16(Aw0, Bx0, acc[0][0]);
        acc[0][1] = wmma_bf16(Aw0, Bx1, acc[0][1]);
        acc[1][0] = wmma_bf16(Aw1, Bx0, acc[1][0]);
        acc[1][1] = wmma_bf16(Aw1, Bx1, acc[1][1]);
      }
    }

  const float gc = *gamma_c_p;
  const float rg = *res_gamma_p;
#pragma unroll
  for (int ps = 0; ps < 2; ++ps) {
    const int p = pb + ps * 16 + l15;
    float vcv[NLAT];
#pragma unroll
    for (int j = 0; j < NLAT; ++j)
      vcv[j] = vc[((size_t)b * NLAT + j) * NPIX + p];
#pragma unroll
    for (int os = 0; os < 2; ++os)
#pragma unroll
      for (int r = 0; r < 8; ++r) {
        const int o = ob + os * 16 + r + 8 * half;
        const size_t idx = ((size_t)b * NC + o) * NPIX + p;
        float chan = bo_c[o];
        const float* Mrow = Mmat + ((size_t)b * NC + o) * NLAT;
#pragma unroll
        for (int j = 0; j < NLAT; ++j) chan = fmaf(Mrow[j], vcv[j], chan);
        float y = acc[os][ps][r];
        y = (y > 0.f) ? y : 0.2f * y;  // lrelu(0.2)
        out[idx] = x[idx] + fmbuf[idx] + gc * chan + rg * y;
      }
  }
}

// ---------------------------------------------------------------------------
extern "C" void kernel_launch(void* const* d_in, const int* in_sizes, int n_in,
                              void* d_out, int out_size, void* d_ws,
                              size_t ws_size, hipStream_t stream) {
  const float* x        = (const float*)d_in[0];
  const float* wq_fm    = (const float*)d_in[1];
  const float* bq_fm    = (const float*)d_in[2];
  const float* wk_fm    = (const float*)d_in[3];
  const float* bk_fm    = (const float*)d_in[4];
  const float* wv_fm    = (const float*)d_in[5];
  const float* bv_fm    = (const float*)d_in[6];
  const float* gamma_fm = (const float*)d_in[7];
  const float* wq_c     = (const float*)d_in[8];
  const float* bq_c     = (const float*)d_in[9];
  const float* wk_c     = (const float*)d_in[10];
  const float* bk_c     = (const float*)d_in[11];
  const float* wv_c     = (const float*)d_in[12];
  const float* bv_c     = (const float*)d_in[13];
  const float* wo_c     = (const float*)d_in[14];
  const float* bo_c     = (const float*)d_in[15];
  const float* gamma_c  = (const float*)d_in[16];
  const float* w_bar    = (const float*)d_in[17];
  const float* u_sn     = (const float*)d_in[18];
  // d_in[19] (v_sn) is unused by the reference math.
  const float* res_gamma = (const float*)d_in[20];
  float* out = (float*)d_out;

  // workspace carve-up (~43.5 MB, 256B aligned)
  size_t off = 0;
  char* base = (char*)d_ws;
  auto alloc = [&](size_t bytes) -> void* {
    void* p = base + off;
    off += (bytes + 255) & ~(size_t)255;
    return p;
  };
  u16*   xTh   = (u16*)alloc((size_t)NB * NPIX * NC * 2);    // 8 MB
  u16*   qfmT  = (u16*)alloc((size_t)NB * NPIX * NLAT * 2);  // 1 MB
  u16*   kfmT  = (u16*)alloc((size_t)NB * NPIX * NLAT * 2);  // 1 MB
  u16*   vfm_h = (u16*)alloc((size_t)NB * NC * NPIX * 2);    // 8 MB
  float* qc_f  = (float*)alloc((size_t)NB * NLAT * NPIX * 4);
  float* kc_f  = (float*)alloc((size_t)NB * NLAT * NPIX * 4);
  float* vc_f  = (float*)alloc((size_t)NB * NLAT * NPIX * 4);
  float* fmbuf = (float*)alloc((size_t)NB * NC * NPIX * 4);  // 16.8 MB
  float* energy = (float*)alloc((size_t)NB * NLAT * NLAT * 4);
  float* Mmat   = (float*)alloc((size_t)NB * NC * NLAT * 4);
  u16*   wh     = (u16*)alloc((size_t)9 * NC * NC * 2);      // 1.2 MB
  float* invsig = (float*)alloc(256);
  (void)ws_size; (void)n_in; (void)in_sizes; (void)out_size;

  k_prep_xT<<<dim3((NB * NC * NPIX) / 256), 256, 0, stream>>>(x, xTh);
  k_qkv<<<dim3(NB * (NPIX / 32)), 256, 0, stream>>>(
      x, wq_fm, bq_fm, wk_fm, bk_fm, wv_fm, bv_fm, wq_c, bq_c, wk_c, bk_c,
      wv_c, bv_c, qfmT, kfmT, vfm_h, qc_f, kc_f, vc_f);
  k_chan_energy<<<dim3(NB, NLAT), 256, 0, stream>>>(kc_f, qc_f, energy);
  k_chan_attn_M<<<dim3(NB), 256, 0, stream>>>(energy, wo_c, Mmat);
  k_sn<<<dim3(1), 256, 0, stream>>>(w_bar, u_sn, invsig);
  k_wconv<<<dim3((9 * NC * NC) / 256), 256, 0, stream>>>(w_bar, invsig, wh);
  k_flash<<<dim3(NPIX / 32, NB), FL_WAVES * 32, 0, stream>>>(
      qfmT, kfmT, vfm_h, gamma_fm, fmbuf);
  k_conv_final<<<dim3(NPIX / 128, NC / 32, NB), 128, 0, stream>>>(
      xTh, wh, x, fmbuf, Mmat, vc_f, bo_c, gamma_c, res_gamma, out);
}
